// Attention_16071767622411
// MI455X (gfx1250) — matrix operand order repaired and checked by reading, not compiled, not run on hardware
//
#include <hip/hip_runtime.h>
#include <hip/hip_bf16.h>
#include <stdint.h>

typedef __attribute__((ext_vector_type(16))) _Float16 v16h;
typedef __attribute__((ext_vector_type(8)))  _Float16 v8h;
typedef __attribute__((ext_vector_type(8)))  float    v8f;

#define B_       8
#define SEQ      1024
#define CH       256
#define HEADS_   8
#define DH_      32
#define HID_     256
#define NOUT_QKV 768
#define QKV_ELEMS (B_ * HEADS_ * SEQ * DH_)   // 2,097,152 per tensor
#define SCALE_ 0.17677669529663687f           // 1/sqrt(32)

__device__ __forceinline__ v8f wmma16x16x32(v16h a, v16h b, v8f c) {
  return __builtin_amdgcn_wmma_f32_16x16x32_f16(false, a, false, b, (short)0, c,
                                                false, false);
}

__device__ __forceinline__ v16h cat8(v8h lo, v8h hi) {
  return __builtin_shufflevector(lo, hi, 0, 1, 2, 3, 4, 5, 6, 7,
                                 8, 9, 10, 11, 12, 13, 14, 15);
}

// 16 contiguous halfs at p (16B aligned) -> fragment elements e=0..15
__device__ __forceinline__ v16h frag_contig(const _Float16* p) {
  return cat8(*(const v8h*)p, *(const v8h*)(p + 8));
}

// A-fragment (16x32 f16, row-major, lda elems): lane holds row m=lane%16;
// e=0..15 -> k=g*16+e  => 16 contiguous halfs
__device__ __forceinline__ v16h load_a_frag16(const _Float16* A, int lda, int lane) {
  const int m = lane & 15, g = lane >> 4;
  const _Float16* p = A + m * lda + (g << 4);
  return cat8(*(const v8h*)p, *(const v8h*)(p + 8));
}

// Same A-fragment but from fp32 row-major source (converts to f16)
__device__ __forceinline__ v16h load_a_frag32(const float* A, int lda, int lane) {
  const int m = lane & 15, g = lane >> 4;
  const float* p = A + m * lda + (g << 4);
  const float4 a0 = ((const float4*)p)[0];
  const float4 a1 = ((const float4*)p)[1];
  const float4 a2 = ((const float4*)(p + 8))[0];
  const float4 a3 = ((const float4*)(p + 8))[1];
  v16h a;
  a[0] = (_Float16)a0.x;  a[1] = (_Float16)a0.y;  a[2] = (_Float16)a0.z;  a[3] = (_Float16)a0.w;
  a[4] = (_Float16)a1.x;  a[5] = (_Float16)a1.y;  a[6] = (_Float16)a1.z;  a[7] = (_Float16)a1.w;
  a[8] = (_Float16)a2.x;  a[9] = (_Float16)a2.y;  a[10] = (_Float16)a2.z; a[11] = (_Float16)a2.w;
  a[12] = (_Float16)a3.x; a[13] = (_Float16)a3.y; a[14] = (_Float16)a3.z; a[15] = (_Float16)a3.w;
  return a;
}

// B-fragment (32x16) from K-contiguous storage M[n][k] (ldk elems per column n):
// lane holds col n=lane%16; e -> k = (lane/16)*16 + e  => 16 contiguous halfs
__device__ __forceinline__ v16h load_b_kcontig(const _Float16* M, int ldk, int lane) {
  return frag_contig(M + (lane & 15) * ldk + ((lane >> 4) << 4));
}

// ---------------------------------------------------------------------------
// Kernel 1: QKV projection  [8192,256] x [256,768] -> head-major f16.
// Q pre-scaled by 1/sqrt(DH); V stored d-major ([b,h,d,n]) so the attention
// O = P*V B-fragments are contiguous loads (free transpose in the scatter).
// ---------------------------------------------------------------------------
__global__ __launch_bounds__(256) void qkv_proj_kernel(
    const float* __restrict__ x, const float* __restrict__ w_qkv,
    _Float16* __restrict__ qkv) {
  __shared__ _Float16 Bsw[64][40];   // B tile, swizzled [n][k] (K-contiguous)
  const int tid = threadIdx.x;
  const int lane = tid & 31, wave = tid >> 5;
  const int m0 = blockIdx.x * 64, n0 = blockIdx.y * 64;
  const int t0 = wave * 2;
  const int tm = t0 >> 2;
  const int tn0 = t0 & 3, tn1 = tn0 + 1;
  const int rb = tid >> 3;              // 0..31 (k row of B tile)
  const int cb0 = (tid & 7) * 8;        // 0..56 (n col start)
  v8f acc0 = {}, acc1 = {};

  for (int k0 = 0; k0 < CH; k0 += 32) {
    __syncthreads();
    {
      const float4* wp = (const float4*)(w_qkv + (k0 + rb) * NOUT_QKV + n0 + cb0);
      const float4 w0 = wp[0], w1 = wp[1];
      Bsw[cb0 + 0][rb] = (_Float16)w0.x;  Bsw[cb0 + 1][rb] = (_Float16)w0.y;
      Bsw[cb0 + 2][rb] = (_Float16)w0.z;  Bsw[cb0 + 3][rb] = (_Float16)w0.w;
      Bsw[cb0 + 4][rb] = (_Float16)w1.x;  Bsw[cb0 + 5][rb] = (_Float16)w1.y;
      Bsw[cb0 + 6][rb] = (_Float16)w1.z;  Bsw[cb0 + 7][rb] = (_Float16)w1.w;
    }
    __syncthreads();
    const v16h a  = load_a_frag32(x + (m0 + tm * 16) * CH + k0, CH, lane);
    const v16h b0 = load_b_kcontig(&Bsw[tn0 * 16][0], 40, lane);
    const v16h b1 = load_b_kcontig(&Bsw[tn1 * 16][0], 40, lane);
    acc0 = wmma16x16x32(a, b0, acc0);
    acc1 = wmma16x16x32(a, b1, acc1);
  }

  const int g = lane >> 4, n = lane & 15;
#pragma unroll
  for (int t = 0; t < 2; ++t) {
    const v8f acc = t ? acc1 : acc0;
    const int tn = t ? tn1 : tn0;
    const int col = n0 + tn * 16 + n;          // 0..767 (uniform 'which' per tile)
    const int which = col >> 8;                // 0=q 1=k 2=v
    const int hid = col & 255;
    const int head = hid >> 5, d = hid & 31;
    const float sc = (which == 0) ? SCALE_ : 1.0f;
    _Float16* dst = qkv + which * QKV_ELEMS;
#pragma unroll
    for (int r = 0; r < 8; ++r) {
      const int row = m0 + tm * 16 + g * 8 + r;   // 0..8191
      const int bb = row >> 10, pos = row & 1023;
      const _Float16 val = (_Float16)(acc[r] * sc);
      if (which == 2)   // V: d-major [b,h,d,n]
        dst[(((bb * HEADS_ + head) * DH_ + d) << 10) + pos] = val;
      else              // Q,K: [b,h,n,d]
        dst[(((bb * HEADS_ + head) * SEQ + pos) << 5) + d] = val;
    }
  }
}

// ---------------------------------------------------------------------------
// Kernel 2: flash attention. One wave per 16 query rows of one (b,head).
// K blocks staged to per-wave LDS via async global->LDS copies (ASYNCcnt);
// V read via contiguous d-major global loads; running-max softmax.
// ---------------------------------------------------------------------------
__global__ __launch_bounds__(128) void attn_kernel(
    const _Float16* __restrict__ qh, const _Float16* __restrict__ kh,
    const _Float16* __restrict__ vt, _Float16* __restrict__ ah) {
  __shared__ _Float16 sK[4][32][32];   // per-wave K block (async copy target)
  __shared__ _Float16 sP[4][16][40];   // per-wave P staging (D-layout -> A-layout)
  const int tid = threadIdx.x;
  const int lane = tid & 31, wave = tid >> 5;
  const int bh = blockIdx.x;               // 0..63
  const int b = bh >> 3, h = bh & 7;
  const int qblk = blockIdx.y * 4 + wave;  // 0..63
  const int g = lane >> 4, n = lane & 15;

  const _Float16* Qb = qh + ((bh * SEQ + qblk * 16) << 5);
  const _Float16* Kb = kh + ((bh * SEQ) << 5);
  const _Float16* Vt = vt + ((size_t)(bh * DH_) << 10);   // [d][n], row = 1024

  const v16h qf = load_a_frag16(Qb, DH_, lane);   // Q already scaled
  v8f o_lo = {}, o_hi = {};
  float mrun[8], lrun[8];
#pragma unroll
  for (int r = 0; r < 8; ++r) { mrun[r] = -1e30f; lrun[r] = 0.0f; }

  const uint32_t klds = (uint32_t)(uintptr_t)&sK[wave][lane][0];

  for (int j0 = 0; j0 < SEQ; j0 += 32) {
    // async copy: lane copies K row (j0+lane), 64B = 4 x b128
    const _Float16* kg = Kb + ((j0 + lane) << 5);
    asm volatile(
        "global_load_async_to_lds_b128 %0, %1, off\n\t"
        "global_load_async_to_lds_b128 %0, %1, off offset:16\n\t"
        "global_load_async_to_lds_b128 %0, %1, off offset:32\n\t"
        "global_load_async_to_lds_b128 %0, %1, off offset:48"
        :: "v"(klds), "v"(kg) : "memory");
    if (j0 + 32 < SEQ) {   // uniform branch: prefetch next K/V blocks
      __builtin_prefetch(Kb + ((j0 + 32 + lane) << 5), 0, 0);
      __builtin_prefetch(Vt + ((size_t)lane << 10) + j0 + 32, 0, 0);
    }
    asm volatile("s_wait_asynccnt 0x0" ::: "memory");

    // S tiles from per-wave LDS K block: B[k=d][n=key] = sK[key][d]
    const v16h kb0 = load_b_kcontig(&sK[wave][0][0], 32, lane);
    const v16h kb1 = load_b_kcontig(&sK[wave][16][0], 32, lane);
    const v8f zero = {};
    const v8f s0 = wmma16x16x32(qf, kb0, zero);
    const v8f s1 = wmma16x16x32(qf, kb1, zero);

#pragma unroll
    for (int r = 0; r < 8; ++r) {
      float mx = fmaxf(s0[r], s1[r]);
#pragma unroll
      for (int off = 1; off < 16; off <<= 1)
        mx = fmaxf(mx, __shfl_xor(mx, off, 32));      // row max (16-lane half)
      const float newm = fmaxf(mrun[r], mx);
      const float alpha = __expf(mrun[r] - newm);
      const float p0 = __expf(s0[r] - newm);
      const float p1 = __expf(s1[r] - newm);
      float sum = p0 + p1;
#pragma unroll
      for (int off = 1; off < 16; off <<= 1)
        sum += __shfl_xor(sum, off, 32);              // row sum
      lrun[r] = lrun[r] * alpha + sum;
      mrun[r] = newm;
      o_lo[r] *= alpha;
      o_hi[r] *= alpha;
      sP[wave][g * 8 + r][n]      = (_Float16)p0;
      sP[wave][g * 8 + r][16 + n] = (_Float16)p1;
    }
    __syncthreads();                                  // uniform across waves
    const v16h pf = load_a_frag16(&sP[wave][0][0], 40, lane);
    __syncthreads();

    // O += P[16x32] * V[32x32]; V is d-major so B-frags are contiguous:
    // B[k=key][n=dcol] = Vt[dcol][j0+key]
    const v16h vb0 = frag_contig(Vt + ((size_t)n << 10) + j0 + (g << 4));
    const v16h vb1 = frag_contig(Vt + ((size_t)(16 + n) << 10) + j0 + (g << 4));
    o_lo = wmma16x16x32(pf, vb0, o_lo);
    o_hi = wmma16x16x32(pf, vb1, o_hi);
  }

#pragma unroll
  for (int r = 0; r < 8; ++r) {
    const int pos = qblk * 16 + g * 8 + r;
    const float inv = 1.0f / lrun[r];
    const int base = (b * SEQ + pos) * HID_ + (h << 5);
    ah[base + n]      = (_Float16)(o_lo[r] * inv);
    ah[base + 16 + n] = (_Float16)(o_hi[r] * inv);
  }
}

// ---------------------------------------------------------------------------
// Kernel 3: output projection [8192,256](f16) x [256,256] -> fp32 d_out
// ---------------------------------------------------------------------------
__global__ __launch_bounds__(256) void out_proj_kernel(
    const _Float16* __restrict__ ah, const float* __restrict__ w_out,
    float* __restrict__ out) {
  __shared__ _Float16 Bsw[64][40];   // [n][k] swizzled weight tile
  const int tid = threadIdx.x;
  const int lane = tid & 31, wave = tid >> 5;
  const int m0 = blockIdx.x * 64, n0 = blockIdx.y * 64;
  const int t0 = wave * 2;
  const int tm = t0 >> 2;
  const int tn0 = t0 & 3, tn1 = tn0 + 1;
  const int rb = tid >> 3;
  const int cb0 = (tid & 7) * 8;
  v8f acc0 = {}, acc1 = {};

  for (int k0 = 0; k0 < HID_; k0 += 32) {
    __syncthreads();
    {
      const float4* wp = (const float4*)(w_out + (k0 + rb) * CH + n0 + cb0);
      const float4 w0 = wp[0], w1 = wp[1];
      Bsw[cb0 + 0][rb] = (_Float16)w0.x;  Bsw[cb0 + 1][rb] = (_Float16)w0.y;
      Bsw[cb0 + 2][rb] = (_Float16)w0.z;  Bsw[cb0 + 3][rb] = (_Float16)w0.w;
      Bsw[cb0 + 4][rb] = (_Float16)w1.x;  Bsw[cb0 + 5][rb] = (_Float16)w1.y;
      Bsw[cb0 + 6][rb] = (_Float16)w1.z;  Bsw[cb0 + 7][rb] = (_Float16)w1.w;
    }
    __syncthreads();
    const v16h a  = load_a_frag16(ah + (m0 + tm * 16) * HID_ + k0, HID_, lane);
    const v16h b0 = load_b_kcontig(&Bsw[tn0 * 16][0], 40, lane);
    const v16h b1 = load_b_kcontig(&Bsw[tn1 * 16][0], 40, lane);
    acc0 = wmma16x16x32(a, b0, acc0);
    acc1 = wmma16x16x32(a, b1, acc1);
  }

  const int g = lane >> 4, n = lane & 15;
#pragma unroll
  for (int t = 0; t < 2; ++t) {
    const v8f acc = t ? acc1 : acc0;
    const int tn = t ? tn1 : tn0;
#pragma unroll
    for (int r = 0; r < 8; ++r)
      out[(m0 + tm * 16 + g * 8 + r) * CH + n0 + tn * 16 + n] = acc[r];
  }
}

extern "C" void kernel_launch(void* const* d_in, const int* in_sizes, int n_in,
                              void* d_out, int out_size, void* d_ws, size_t ws_size,
                              hipStream_t stream) {
  const float* x     = (const float*)d_in[0];   // [8,32,32,256]
  const float* w_qkv = (const float*)d_in[1];   // [256,768]
  const float* w_out = (const float*)d_in[2];   // [256,256]
  float* out = (float*)d_out;                   // [8,32,32,256] fp32

  _Float16* qh = (_Float16*)d_ws;               // [b,h,n,d] f16, pre-scaled
  _Float16* kh = qh + QKV_ELEMS;                // [b,h,n,d] f16
  _Float16* vt = kh + QKV_ELEMS;                // [b,h,d,n] f16 (transposed)
  _Float16* ah = vt + QKV_ELEMS;                // [b,n,hid] f16

  qkv_proj_kernel<<<dim3(128, 12), 256, 0, stream>>>(x, w_qkv, qh);
  attn_kernel   <<<dim3(64, 16),  128, 0, stream>>>(qh, kh, vt, ah);
  out_proj_kernel<<<dim3(128, 4), 256, 0, stream>>>(ah, w_out, out);
}
